// GatedGCNNet_Edge_57775900066614
// MI455X (gfx1250) — matrix operand order repaired and checked by reading, not compiled, not run on hardware
//
#include <hip/hip_runtime.h>
#include <hip/hip_fp16.h>

// ---------------------------------------------------------------------------
// GatedGCN (edge output) for MI455X / gfx1250, wave32 + WMMA f16->f32.
// LDS operands pre-swizzled into WMMA fragment layout -> vector ds loads.
// ---------------------------------------------------------------------------

typedef __attribute__((ext_vector_type(16))) _Float16 v16h;
typedef __attribute__((ext_vector_type(8)))  float    v8f;

#define BN_EPS 1e-5f

// Convert 8 consecutive f32 to halves [base, base+8) of a v16h A-tile fragment.
__device__ __forceinline__ void cvt8(const float* __restrict__ p, v16h& a, int base) {
  const float4 q0 = *reinterpret_cast<const float4*>(p);
  const float4 q1 = *reinterpret_cast<const float4*>(p + 4);
  a[base + 0] = (_Float16)q0.x; a[base + 1] = (_Float16)q0.y;
  a[base + 2] = (_Float16)q0.z; a[base + 3] = (_Float16)q0.w;
  a[base + 4] = (_Float16)q1.x; a[base + 5] = (_Float16)q1.y;
  a[base + 6] = (_Float16)q1.z; a[base + 7] = (_Float16)q1.w;
}

__device__ __forceinline__ v16h lds_frag(const _Float16* p) {
  return *reinterpret_cast<const v16h*>(p);            // 32B aligned -> ds_load_b128 x2
}

// ---------------------------------------------------------------------------
// Y[rows,64] = X[rows,64] @ W[64,64] + bias    (WMMA f16 in, f32 accumulate)
// Block = 256 threads = 8 waves; each wave computes a 16-row stripe.
// W held in LDS pre-swizzled as [tile=nt*2+kk][lane][16] fragments.
// ---------------------------------------------------------------------------
__global__ __launch_bounds__(256) void gemm_x64_wmma(
    const float* __restrict__ X, const float* __restrict__ W,
    const float* __restrict__ bias, float* __restrict__ Y, int rows)
{
  __shared__ __align__(32) _Float16 Ws[8 * 512];       // 8 KB, B-fragment layout
  for (int idx = threadIdx.x; idx < 8 * 512; idx += 256) {
    const int t  = idx >> 9;            // tile: nt*2+kk
    const int ln = (idx >> 4) & 31;
    const int i  = idx & 15;
    const int nt = t >> 1, kk = t & 1;
    const int k  = kk * 32 + ((i >> 3) << 4) + ((ln >> 4) << 3) + (i & 7);
    const int n  = nt * 16 + (ln & 15);
    Ws[idx] = (_Float16)W[k * 64 + n];
  }
  __syncthreads();

  const int wave = threadIdx.x >> 5;
  const int lane = threadIdx.x & 31;
  const int r0 = (blockIdx.x * 8 + wave) * 16;
  if (r0 >= rows) return;                              // whole-wave uniform
  const int lrow = lane & 15;
  const int lhi  = lane >> 4;
  int row = r0 + lrow; if (row > rows - 1) row = rows - 1;   // clamp tail loads
  const float* xrow = X + (size_t)row * 64;

  // A: 16x32 f16 tiles, lane l holds row (l&15); halves 0-7 -> K=kb0.., 8-15 -> K=kb0+16..
  v16h a[2];
  #pragma unroll
  for (int kk = 0; kk < 2; ++kk) {
    const int kb0 = kk * 32 + lhi * 8;
    cvt8(xrow + kb0,      a[kk], 0);
    cvt8(xrow + kb0 + 16, a[kk], 8);
  }

  // Preload all 8 B fragments (one aligned 32B LDS load each).
  v16h b[8];
  #pragma unroll
  for (int t = 0; t < 8; ++t) b[t] = lds_frag(Ws + t * 512 + lane * 16);

  v8f c[4] = {};
  #pragma unroll
  for (int nt = 0; nt < 4; ++nt)
    #pragma unroll
    for (int kk = 0; kk < 2; ++kk)
      c[nt] = __builtin_amdgcn_wmma_f32_16x16x32_f16(
          false, a[kk], false, b[nt * 2 + kk], (short)0, c[nt], false, false);

  #pragma unroll
  for (int nt = 0; nt < 4; ++nt) {
    const int n = nt * 16 + lrow;
    const float bv = bias[n];
    #pragma unroll
    for (int j = 0; j < 8; ++j) {
      const int m = r0 + j + 8 * lhi;
      if (m < rows) Y[(size_t)m * 64 + n] = c[nt][j] + bv;
    }
  }
}

// D-fragment element (m, n) -> A-fragment staging offset (within one 16xK tile set).
// A layout: (lane, i) holds X[lane&15][kk*32 + (i>>3)*16 + (lane>>4)*8 + (i&7)].
__device__ __forceinline__ int a_swz(int m, int n) {
  const int kk   = n >> 5;
  const int ln   = m + 16 * ((n >> 3) & 1);
  const int i    = ((n >> 4) & 1) * 8 + (n & 7);
  return kk * 512 + ln * 16 + i;
}

// ---------------------------------------------------------------------------
// Fused edge readout: x = concat(h[src],h[dst]) [E,128]
//   relu(x@W0+b0) -> relu(@W1+b1) -> @W2+b2 -> out [E,32]
// Each wave processes 16 edges; layer outputs staged in LDS pre-swizzled to
// the next layer's A-fragment layout (vector reload, no scalar gathers).
// ---------------------------------------------------------------------------
__global__ __launch_bounds__(256) void readout_mlp_wmma(
    const float* __restrict__ h, const int* __restrict__ src, const int* __restrict__ dst,
    const float* __restrict__ w0, const float* __restrict__ b0,
    const float* __restrict__ w1, const float* __restrict__ b1,
    const float* __restrict__ w2, const float* __restrict__ b2,
    float* __restrict__ out, int E)
{
  __shared__ __align__(32) _Float16 W0s[16 * 512];     // 16 KB, tiles nt*4+kk
  __shared__ __align__(32) _Float16 W1s[4 * 512];      //  4 KB, tiles nt*2+kk
  __shared__ __align__(32) _Float16 W2s[2 * 512];      //  2 KB, tiles nt
  __shared__ __align__(32) _Float16 stage[8 * 1024];   // 16 KB, per-wave A-swizzled
  for (int idx = threadIdx.x; idx < 16 * 512; idx += 256) {
    const int t = idx >> 9, ln = (idx >> 4) & 31, i = idx & 15;
    const int nt = t >> 2, kk = t & 3;
    const int k = kk * 32 + ((i >> 3) << 4) + ((ln >> 4) << 3) + (i & 7);
    const int n = nt * 16 + (ln & 15);
    W0s[idx] = (_Float16)w0[k * 64 + n];
  }
  for (int idx = threadIdx.x; idx < 4 * 512; idx += 256) {
    const int t = idx >> 9, ln = (idx >> 4) & 31, i = idx & 15;
    const int nt = t >> 1, kk = t & 1;
    const int k = kk * 32 + ((i >> 3) << 4) + ((ln >> 4) << 3) + (i & 7);
    const int n = nt * 16 + (ln & 15);
    W1s[idx] = (_Float16)w1[k * 32 + n];
  }
  for (int idx = threadIdx.x; idx < 2 * 512; idx += 256) {
    const int t = idx >> 9, ln = (idx >> 4) & 31, i = idx & 15;
    const int k = ((i >> 3) << 4) + ((ln >> 4) << 3) + (i & 7);
    const int n = t * 16 + (ln & 15);
    W2s[idx] = (_Float16)w2[k * 32 + n];
  }
  __syncthreads();

  const int wave = threadIdx.x >> 5;
  const int lane = threadIdx.x & 31;
  const int e0 = (blockIdx.x * 8 + wave) * 16;
  if (e0 >= E) return;
  const int lrow = lane & 15;
  const int lhi  = lane >> 4;
  int er = e0 + lrow; if (er > E - 1) er = E - 1;
  const float* hs = h + (size_t)src[er] * 64;
  const float* hd = h + (size_t)dst[er] * 64;
  _Float16* st = stage + wave * 1024;

  // ---- layer 0: [16x128] @ [128x64] ----
  v16h a0[4];
  #pragma unroll
  for (int kk = 0; kk < 4; ++kk) {
    const int kb0 = kk * 32 + lhi * 8;                 // 8-aligned, never straddles 64
    const int kb1 = kb0 + 16;
    const float* p0 = (kb0 < 64) ? (hs + kb0) : (hd + kb0 - 64);
    const float* p1 = (kb1 < 64) ? (hs + kb1) : (hd + kb1 - 64);
    cvt8(p0, a0[kk], 0);
    cvt8(p1, a0[kk], 8);
  }
  v8f c0[4] = {};
  #pragma unroll
  for (int nt = 0; nt < 4; ++nt)
    #pragma unroll
    for (int kk = 0; kk < 4; ++kk) {
      const v16h b = lds_frag(W0s + (nt * 4 + kk) * 512 + lane * 16);
      c0[nt] = __builtin_amdgcn_wmma_f32_16x16x32_f16(
          false, a0[kk], false, b, (short)0, c0[nt], false, false);
    }
  // relu+bias, store directly into next layer's A-fragment positions
  #pragma unroll
  for (int nt = 0; nt < 4; ++nt) {
    const int n = nt * 16 + lrow;
    const float bv = b0[n];
    #pragma unroll
    for (int j = 0; j < 8; ++j)
      st[a_swz(j + 8 * lhi, n)] = (_Float16)fmaxf(c0[nt][j] + bv, 0.f);
  }
  asm volatile("s_wait_dscnt 0" ::: "memory");         // per-wave LDS relayout fence

  // ---- layer 1: [16x64] @ [64x32] ----
  v16h a1[2];
  #pragma unroll
  for (int kk = 0; kk < 2; ++kk) a1[kk] = lds_frag(st + kk * 512 + lane * 16);
  v8f c1[2] = {};
  #pragma unroll
  for (int nt = 0; nt < 2; ++nt)
    #pragma unroll
    for (int kk = 0; kk < 2; ++kk) {
      const v16h b = lds_frag(W1s + (nt * 2 + kk) * 512 + lane * 16);
      c1[nt] = __builtin_amdgcn_wmma_f32_16x16x32_f16(
          false, a1[kk], false, b, (short)0, c1[nt], false, false);
    }
  asm volatile("s_wait_dscnt 0" ::: "memory");
  #pragma unroll
  for (int nt = 0; nt < 2; ++nt) {
    const int n = nt * 16 + lrow;
    const float bv = b1[n];
    #pragma unroll
    for (int j = 0; j < 8; ++j)
      st[a_swz(j + 8 * lhi, n)] = (_Float16)fmaxf(c1[nt][j] + bv, 0.f);
  }
  asm volatile("s_wait_dscnt 0" ::: "memory");

  // ---- layer 2: [16x32] @ [32x32] ----
  const v16h a2 = lds_frag(st + lane * 16);
  v8f c2[2] = {};
  #pragma unroll
  for (int nt = 0; nt < 2; ++nt) {
    const v16h b = lds_frag(W2s + nt * 512 + lane * 16);
    c2[nt] = __builtin_amdgcn_wmma_f32_16x16x32_f16(
        false, a2, false, b, (short)0, c2[nt], false, false);
  }
  #pragma unroll
  for (int nt = 0; nt < 2; ++nt) {
    const int n = nt * 16 + lrow;
    const float bv = b2[n];
    #pragma unroll
    for (int j = 0; j < 8; ++j) {
      const int m = e0 + j + 8 * lhi;
      if (m < E) out[(size_t)m * 32 + n] = c2[nt][j] + bv;
    }
  }
}

// ---------------------------------------------------------------------------
// Elementwise / scatter / batchnorm helper kernels
// ---------------------------------------------------------------------------
__global__ __launch_bounds__(256) void zero_f32(float* __restrict__ p, long n) {
  long i = (long)blockIdx.x * blockDim.x + threadIdx.x;
  const long st = (long)gridDim.x * blockDim.x;
  for (; i < n; i += st) p[i] = 0.f;
}

// out[r,c] = f[r]*w[c] + b[c]   (rank-1 embedding), n = rows*64
__global__ __launch_bounds__(256) void embed_rank1(
    const float* __restrict__ f, const float* __restrict__ w,
    const float* __restrict__ b, float* __restrict__ out, int n)
{
  const int idx = blockIdx.x * 256 + threadIdx.x;
  if (idx >= n) return;
  out[idx] = f[idx >> 6] * w[idx & 63] + b[idx & 63];
}

// T[dst[e],:] += Bh[src[e],:]   (2 feats / thread, HW f32 atomics)
__global__ __launch_bounds__(256) void scatter_add_iso(
    const float* __restrict__ Bh, const int* __restrict__ src,
    const int* __restrict__ dst, float* __restrict__ T, int E)
{
  const long idx = (long)blockIdx.x * 256 + threadIdx.x;
  if (idx >= (long)E * 32) return;
  const int e = (int)(idx >> 5);
  const int f = ((int)idx & 31) * 2;
  const size_t so  = (size_t)src[e] * 64 + f;
  const size_t dof = (size_t)dst[e] * 64 + f;
  unsafeAtomicAdd(&T[dof],     Bh[so]);
  unsafeAtomicAdd(&T[dof + 1], Bh[so + 1]);
}

// uv[0:64]=emb_e_w@gC ; uv[64:128]=emb_e_b@gC + gC_b   (rank-1 Ce factorization)
__global__ void rank1_ce(const float* __restrict__ we, const float* __restrict__ be,
                         const float* __restrict__ gCw, const float* __restrict__ gCb,
                         float* __restrict__ uv)
{
  const int n = threadIdx.x;
  float u = 0.f, v = 0.f;
  for (int k = 0; k < 64; ++k) { const float w = gCw[k * 64 + n]; u += we[k] * w; v += be[k] * w; }
  uv[n] = u; uv[64 + n] = v + gCb[n];
}

// e_new = ef*u + v + Dh[src] + Eh[dst]; sigma=sigmoid(e_new);
// SSH[dst] += Bh[src]*sigma ; SS[dst] += sigma
__global__ __launch_bounds__(256) void edge_gated(
    const float* __restrict__ ef, const int* __restrict__ src, const int* __restrict__ dst,
    const float* __restrict__ uv, const float* __restrict__ Dh, const float* __restrict__ Eh,
    const float* __restrict__ Bh, float* __restrict__ SSH, float* __restrict__ SS, int E)
{
  const long idx = (long)blockIdx.x * 256 + threadIdx.x;
  if (idx >= (long)E * 32) return;
  const int e = (int)(idx >> 5);
  const int f = ((int)idx & 31) * 2;
  const int s = src[e], d = dst[e];
  const float x = ef[e];
  #pragma unroll
  for (int t = 0; t < 2; ++t) {
    const int ff = f + t;
    const float en = x * uv[ff] + uv[64 + ff] + Dh[(size_t)s * 64 + ff] + Eh[(size_t)d * 64 + ff];
    const float sg = 1.0f / (1.0f + __expf(-en));
    unsafeAtomicAdd(&SSH[(size_t)d * 64 + ff], Bh[(size_t)s * 64 + ff] * sg);
    unsafeAtomicAdd(&SS[(size_t)d * 64 + ff], sg);
  }
}

__global__ __launch_bounds__(256) void combine_gated(
    float* __restrict__ T, const float* __restrict__ SSH,
    const float* __restrict__ SS, int n)
{
  const int idx = blockIdx.x * 256 + threadIdx.x;
  if (idx >= n) return;
  T[idx] += SSH[idx] / (SS[idx] + 1e-6f);
}

// Per-column sum / sumsq partials -> f32 atomics into accum[0:64]/[64:128]
__global__ __launch_bounds__(256) void bn_stats(
    const float* __restrict__ X, int rows, float* __restrict__ accum)
{
  __shared__ float ssum[256], ssq[256];
  const int c  = threadIdx.x & 63;
  const int rl = threadIdx.x >> 6;
  const int rbeg = blockIdx.x * 512;
  int rend = rbeg + 512; if (rend > rows) rend = rows;
  float s = 0.f, q = 0.f;
  for (int r = rbeg + rl; r < rend; r += 4) {
    const float x = X[(size_t)r * 64 + c];
    s += x; q += x * x;
  }
  ssum[threadIdx.x] = s; ssq[threadIdx.x] = q;
  __syncthreads();
  if (threadIdx.x < 64) {
    const float S = ssum[c] + ssum[c + 64] + ssum[c + 128] + ssum[c + 192];
    const float Q = ssq[c]  + ssq[c + 64]  + ssq[c + 128]  + ssq[c + 192];
    unsafeAtomicAdd(&accum[c], S);
    unsafeAtomicAdd(&accum[64 + c], Q);
  }
}

// Fold BN to per-column affine: scale = g*rsqrt(var+eps), shift = b - scale*mu
__global__ void bn_finalize(const float* __restrict__ accum,
                            const float* __restrict__ g, const float* __restrict__ b,
                            float inv_rows, float* __restrict__ scsh)
{
  const int c = threadIdx.x;
  const float mu  = accum[c] * inv_rows;
  const float var = accum[64 + c] * inv_rows - mu * mu;
  const float sc  = g[c] * rsqrtf(var + BN_EPS);
  scsh[c] = sc;
  scsh[64 + c] = b[c] - sc * mu;
}

// h += relu(scale*T + shift)
__global__ __launch_bounds__(256) void bn_apply_res(
    const float* __restrict__ T, const float* __restrict__ scsh,
    float* __restrict__ h, int n)
{
  const int idx = blockIdx.x * 256 + threadIdx.x;
  if (idx >= n) return;
  const int c = idx & 63;
  h[idx] += fmaxf(scsh[c] * T[idx] + scsh[64 + c], 0.f);
}

// ---------------------------------------------------------------------------
extern "C" void kernel_launch(void* const* d_in, const int* in_sizes, int n_in,
                              void* d_out, int out_size, void* d_ws, size_t ws_size,
                              hipStream_t stream)
{
  (void)n_in; (void)out_size; (void)ws_size;
  const float* node_feat = (const float*)d_in[0];
  const float* edge_feat = (const float*)d_in[1];
  const int*   src       = (const int*)d_in[2];
  const int*   dst       = (const int*)d_in[3];
  const float* emb_h_w = (const float*)d_in[4];
  const float* emb_h_b = (const float*)d_in[5];
  const float* emb_e_w = (const float*)d_in[6];
  const float* emb_e_b = (const float*)d_in[7];
  const float* iso_A_w = (const float*)d_in[8];
  const float* iso_A_b = (const float*)d_in[9];
  const float* iso_B_w = (const float*)d_in[10];
  const float* iso_B_b = (const float*)d_in[11];
  const float* iso_bn_g = (const float*)d_in[12];
  const float* iso_bn_b = (const float*)d_in[13];
  const float* gA_w = (const float*)d_in[14];
  const float* gA_b = (const float*)d_in[15];
  const float* gB_w = (const float*)d_in[16];
  const float* gB_b = (const float*)d_in[17];
  const float* gC_w = (const float*)d_in[18];
  const float* gC_b = (const float*)d_in[19];
  const float* gD_w = (const float*)d_in[20];
  const float* gD_b = (const float*)d_in[21];
  const float* gE_w = (const float*)d_in[22];
  const float* gE_b = (const float*)d_in[23];
  const float* bn_h_g = (const float*)d_in[24];
  const float* bn_h_b = (const float*)d_in[25];
  // d_in[26], d_in[27] (bn_e) are dead w.r.t. the output; skipped.
  const float* mlp_w0 = (const float*)d_in[28];
  const float* mlp_b0 = (const float*)d_in[29];
  const float* mlp_w1 = (const float*)d_in[30];
  const float* mlp_b1 = (const float*)d_in[31];
  const float* mlp_w2 = (const float*)d_in[32];
  const float* mlp_b2 = (const float*)d_in[33];

  const int N = in_sizes[0];          // node_feat is [N,1]
  const int E = in_sizes[2];          // src is [E]
  const size_t N64 = (size_t)N * 64;

  float* ws  = (float*)d_ws;
  float* h   = ws; ws += N64;
  float* T   = ws; ws += N64;
  float* Bh  = ws; ws += N64;
  float* Dh  = ws; ws += N64;
  float* Eh  = ws; ws += N64;
  float* SSH = ws; ws += N64;          // SSH and SS contiguous (zeroed together)
  float* SS  = ws; ws += N64;
  float* accum = ws; ws += 128;
  float* scsh  = ws; ws += 128;
  float* uv    = ws; ws += 128;

  const int elemBlocksN = (int)((N64 + 255) / 256);
  const int gemmBlocksN = (N + 127) / 128;
  const int edgeBlocks  = (int)(((size_t)E * 32 + 255) / 256);
  const int statBlocksN = (N + 511) / 512;
  const float invN = 1.0f / (float)N;

  embed_rank1<<<elemBlocksN, 256, 0, stream>>>(node_feat, emb_h_w, emb_h_b, h, (int)N64);

  for (int i = 0; i < 3; ++i) {
    gemm_x64_wmma<<<gemmBlocksN, 256, 0, stream>>>(h, iso_A_w + i * 4096, iso_A_b + i * 64, T, N);
    gemm_x64_wmma<<<gemmBlocksN, 256, 0, stream>>>(h, iso_B_w + i * 4096, iso_B_b + i * 64, Bh, N);
    scatter_add_iso<<<edgeBlocks, 256, 0, stream>>>(Bh, src, dst, T, E);
    zero_f32<<<1, 128, 0, stream>>>(accum, 128);
    bn_stats<<<statBlocksN, 256, 0, stream>>>(T, N, accum);
    bn_finalize<<<1, 64, 0, stream>>>(accum, iso_bn_g + i * 64, iso_bn_b + i * 64, invN, scsh);
    bn_apply_res<<<elemBlocksN, 256, 0, stream>>>(T, scsh, h, (int)N64);
  }

  // gated layer
  gemm_x64_wmma<<<gemmBlocksN, 256, 0, stream>>>(h, gA_w, gA_b, T, N);
  gemm_x64_wmma<<<gemmBlocksN, 256, 0, stream>>>(h, gB_w, gB_b, Bh, N);
  gemm_x64_wmma<<<gemmBlocksN, 256, 0, stream>>>(h, gD_w, gD_b, Dh, N);
  gemm_x64_wmma<<<gemmBlocksN, 256, 0, stream>>>(h, gE_w, gE_b, Eh, N);
  rank1_ce<<<1, 64, 0, stream>>>(emb_e_w, emb_e_b, gC_w, gC_b, uv);
  zero_f32<<<4096, 256, 0, stream>>>(SSH, (long)(2 * N64));
  edge_gated<<<edgeBlocks, 256, 0, stream>>>(edge_feat, src, dst, uv, Dh, Eh, Bh, SSH, SS, E);
  combine_gated<<<elemBlocksN, 256, 0, stream>>>(T, SSH, SS, (int)N64);
  zero_f32<<<1, 128, 0, stream>>>(accum, 128);
  bn_stats<<<statBlocksN, 256, 0, stream>>>(T, N, accum);
  bn_finalize<<<1, 64, 0, stream>>>(accum, bn_h_g, bn_h_b, invN, scsh);
  bn_apply_res<<<elemBlocksN, 256, 0, stream>>>(T, scsh, h, (int)N64);

  // fused MLP readout on edges
  const int roBlocks = (E + 127) / 128;
  readout_mlp_wmma<<<roBlocks, 256, 0, stream>>>(h, src, dst,
      mlp_w0, mlp_b0, mlp_w1, mlp_b1, mlp_w2, mlp_b2, (float*)d_out, E);
}
